// Model_structure_14998025798311
// MI455X (gfx1250) — matrix operand order repaired and checked by reading, not compiled, not run on hardware
//
#include <hip/hip_runtime.h>

typedef __attribute__((ext_vector_type(16))) _Float16 v16h;
typedef __attribute__((ext_vector_type(8)))  _Float16 v8h;
typedef __attribute__((ext_vector_type(8)))  float    v8f;

#define BDIM   4096
#define DDIM   1024
#define KSEL   1024
#define MARGINF 0.5f
#define EPSF    1e-6f
#define BIGF    1.0e6f

// ---------------------------------------------------------------------------
// Kernel 1: split x+eps and y into f16 hi/lo pairs, compute squared norms and
// the exact-f32 diagonal distance pos[i] = ||(x_i+eps) - y_i||.
// ---------------------------------------------------------------------------
__global__ void __launch_bounds__(256)
prep_kernel(const float* __restrict__ x, const float* __restrict__ y,
            _Float16* __restrict__ xh, _Float16* __restrict__ xl,
            _Float16* __restrict__ yh, _Float16* __restrict__ yl,
            float* __restrict__ xnorm, float* __restrict__ ynorm,
            float* __restrict__ pos)
{
    const int row = blockIdx.x;
    const int t   = threadIdx.x;
    __shared__ float sx[256], sy[256], sxy[256];

    float ax = 0.f, ay = 0.f, axy = 0.f;
    const size_t base = (size_t)row * DDIM;
    for (int d = t; d < DDIM; d += 256) {
        float xv = x[base + d] + EPSF;
        float yv = y[base + d];
        _Float16 xhv = (_Float16)xv;
        _Float16 yhv = (_Float16)yv;
        xh[base + d] = xhv;
        xl[base + d] = (_Float16)(xv - (float)xhv);
        yh[base + d] = yhv;
        yl[base + d] = (_Float16)(yv - (float)yhv);
        ax  += xv * xv;
        ay  += yv * yv;
        axy += xv * yv;
    }
    sx[t] = ax; sy[t] = ay; sxy[t] = axy;
    __syncthreads();
    for (int off = 128; off > 0; off >>= 1) {
        if (t < off) { sx[t] += sx[t+off]; sy[t] += sy[t+off]; sxy[t] += sxy[t+off]; }
        __syncthreads();
    }
    if (t == 0) {
        xnorm[row] = sx[0];
        ynorm[row] = sy[0];
        float sq = sx[0] + sy[0] - 2.0f * sxy[0];
        pos[row] = sqrtf(fmaxf(sq, 0.0f));
    }
}

// ---------------------------------------------------------------------------
// Kernel 2: distance-matrix GEMM via v_wmma_f32_16x16x32_f16 with a 3-term
// hi/lo split for ~f32 accuracy. Each wave computes a 64x64 output tile;
// block = 8 waves = 128x256 tile. Epilogue forms sqrt(|a|^2+|b|^2-2ab),
// diagonal forced to BIG.
// ---------------------------------------------------------------------------
__device__ inline v16h load_frag16(const _Float16* p0, const _Float16* p1) {
    v8h lo = *(const v8h*)p0;
    v8h hi = *(const v8h*)p1;
    v16h r;
#pragma unroll
    for (int i = 0; i < 8; i++) { r[i] = lo[i]; r[8 + i] = hi[i]; }
    return r;
}

__global__ void __launch_bounds__(256)
dist_gemm(const _Float16* __restrict__ Ah, const _Float16* __restrict__ Al,
          const _Float16* __restrict__ Bh, const _Float16* __restrict__ Bl,
          const float* __restrict__ nA, const float* __restrict__ nB,
          float* __restrict__ out)
{
    const int lane = threadIdx.x & 31;
    const int wv   = threadIdx.x >> 5;     // 0..7
    const int lm   = lane & 15;
    const int lh   = lane >> 4;            // 0/1 lane-half
    const int row0 = blockIdx.y * 128 + (wv >> 2) * 64;
    const int col0 = blockIdx.x * 256 + (wv & 3) * 64;

    v8f acc[4][4];
#pragma unroll
    for (int r = 0; r < 4; r++)
#pragma unroll
        for (int c = 0; c < 4; c++)
#pragma unroll
            for (int i = 0; i < 8; i++) acc[r][c][i] = 0.0f;

    for (int k0 = 0; k0 < DDIM; k0 += 32) {
        // B fragments (K x N = 32 x 16): lane l -> column lm, K halves
        // [k0 + 16*lh .. +15] contiguous (same k order as the A fragments).
        v16h bh[4], bl[4];
#pragma unroll
        for (int c = 0; c < 4; c++) {
            const size_t rb = (size_t)(col0 + c * 16 + lm) * DDIM + k0 + lh * 16;
            bh[c] = load_frag16(Bh + rb, Bh + rb + 8);
            bl[c] = load_frag16(Bl + rb, Bl + rb + 8);
        }
#pragma unroll
        for (int r = 0; r < 4; r++) {
            // A fragments (M x K = 16 x 32): lane l -> row lm, K halves
            // [k0 + 16*lh .. +15] contiguous (matches B layout).
            const size_t ra = (size_t)(row0 + r * 16 + lm) * DDIM + k0 + lh * 16;
            v16h ah = load_frag16(Ah + ra, Ah + ra + 8);
            v16h al = load_frag16(Al + ra, Al + ra + 8);
#pragma unroll
            for (int c = 0; c < 4; c++) {
                acc[r][c] = __builtin_amdgcn_wmma_f32_16x16x32_f16(
                    false, ah, false, bh[c], (short)0, acc[r][c], false, false);
                acc[r][c] = __builtin_amdgcn_wmma_f32_16x16x32_f16(
                    false, ah, false, bl[c], (short)0, acc[r][c], false, false);
                acc[r][c] = __builtin_amdgcn_wmma_f32_16x16x32_f16(
                    false, al, false, bh[c], (short)0, acc[r][c], false, false);
            }
        }
    }

    // Epilogue: C/D layout — VGPR g: lanes 0-15 -> M=g, lanes 16-31 -> M=g+8.
#pragma unroll
    for (int r = 0; r < 4; r++) {
#pragma unroll
        for (int c = 0; c < 4; c++) {
            const int gcol = col0 + c * 16 + lm;
            const float nb = nB[gcol];
#pragma unroll
            for (int g = 0; g < 8; g++) {
                const int grow = row0 + r * 16 + lh * 8 + g;
                float sq = nA[grow] + nb - 2.0f * acc[r][c][g];
                float dv = sqrtf(fmaxf(sq, 0.0f));
                if (grow == gcol) dv = BIGF;
                out[(size_t)grow * BDIM + gcol] = dv;
            }
        }
    }
}

// ---------------------------------------------------------------------------
// Kernel 3: per-row K-th-smallest radix select (on f32 bit pattern; all values
// >= 0 so uint order == float order) + hinge sum over the K selected values.
// One block per row; row staged in LDS (16 KB of the 320 KB/WGP).
// ---------------------------------------------------------------------------
__global__ void __launch_bounds__(256)
topk_hinge_row(const float* __restrict__ mat, const float* __restrict__ pos,
               float* __restrict__ outsum)
{
    const int row = blockIdx.x;
    const int t   = threadIdx.x;
    __shared__ unsigned vals[BDIM];
    __shared__ unsigned hist[256];
    __shared__ unsigned sh_prefix, sh_k;
    __shared__ float red[256];

    const float* rp = mat + (size_t)row * BDIM;
#pragma unroll
    for (int j = 0; j < BDIM / 256; j++)
        vals[t + j * 256] = __float_as_uint(rp[t + j * 256]);
    if (t == 0) { sh_k = KSEL; sh_prefix = 0u; }
    __syncthreads();

    for (int pass = 0; pass < 4; pass++) {
        const int shift = 24 - 8 * pass;
        const unsigned pmask = (pass == 0) ? 0u : (0xFFFFFFFFu << (shift + 8));
        const unsigned pref  = sh_prefix;
        const unsigned kk    = sh_k;
        hist[t] = 0u;
        __syncthreads();
        for (int j = 0; j < BDIM / 256; j++) {
            unsigned u = vals[t + j * 256];
            if ((u & pmask) == pref) atomicAdd(&hist[(u >> shift) & 255u], 1u);
        }
        __syncthreads();
        // inclusive scan over 256 bins (Hillis-Steele)
        for (int off = 1; off < 256; off <<= 1) {
            unsigned add = (t >= off) ? hist[t - off] : 0u;
            __syncthreads();
            hist[t] += add;
            __syncthreads();
        }
        unsigned below = (t == 0) ? 0u : hist[t - 1];
        if (hist[t] >= kk && below < kk) {       // exactly one thread
            sh_prefix = pref | ((unsigned)t << shift);
            sh_k = kk - below;
        }
        __syncthreads();
    }

    const unsigned T    = sh_prefix;   // bit pattern of K-th smallest value
    const unsigned kfin = sh_k;        // copies of T inside the selection
    const float    p    = pos[row];

    float s = 0.f;
    for (int j = 0; j < BDIM / 256; j++) {
        unsigned u = vals[t + j * 256];
        if (u < T) s += fmaxf(MARGINF + p - __uint_as_float(u), 0.0f);
    }
    red[t] = s;
    __syncthreads();
    for (int off = 128; off > 0; off >>= 1) {
        if (t < off) red[t] += red[t + off];
        __syncthreads();
    }
    if (t == 0) {
        float ft = fmaxf(MARGINF + p - __uint_as_float(T), 0.0f);
        outsum[row] = red[0] + (float)kfin * ft;
    }
}

// ---------------------------------------------------------------------------
// Kernel 4: deterministic final reduction (double accumulation, fixed order).
// ---------------------------------------------------------------------------
__global__ void __launch_bounds__(256)
finalize(const float* __restrict__ rs, const float* __restrict__ cs,
         float* __restrict__ out)
{
    const int t = threadIdx.x;
    __shared__ double ra[256], rb[256];
    double a = 0.0, b = 0.0;
    for (int i = t; i < BDIM; i += 256) { a += (double)rs[i]; b += (double)cs[i]; }
    ra[t] = a; rb[t] = b;
    __syncthreads();
    for (int off = 128; off > 0; off >>= 1) {
        if (t < off) { ra[t] += ra[t + off]; rb[t] += rb[t + off]; }
        __syncthreads();
    }
    if (t == 0) {
        const double denom = (double)BDIM * (double)KSEL;
        out[0] = (float)(ra[0] / denom);
        out[1] = (float)(rb[0] / denom);
    }
}

// ---------------------------------------------------------------------------
extern "C" void kernel_launch(void* const* d_in, const int* in_sizes, int n_in,
                              void* d_out, int out_size, void* d_ws, size_t ws_size,
                              hipStream_t stream)
{
    (void)in_sizes; (void)n_in; (void)out_size; (void)ws_size;
    const float* x = (const float*)d_in[0];
    const float* y = (const float*)d_in[1];
    float* out = (float*)d_out;

    char* ws = (char*)d_ws;
    size_t off = 0;
    float*    dist  = (float*)(ws + off);    off += (size_t)BDIM * BDIM * sizeof(float);   // 64 MB
    _Float16* xh    = (_Float16*)(ws + off); off += (size_t)BDIM * DDIM * sizeof(_Float16);
    _Float16* xl    = (_Float16*)(ws + off); off += (size_t)BDIM * DDIM * sizeof(_Float16);
    _Float16* yh    = (_Float16*)(ws + off); off += (size_t)BDIM * DDIM * sizeof(_Float16);
    _Float16* yl    = (_Float16*)(ws + off); off += (size_t)BDIM * DDIM * sizeof(_Float16);
    float*    xnorm = (float*)(ws + off);    off += BDIM * sizeof(float);
    float*    ynorm = (float*)(ws + off);    off += BDIM * sizeof(float);
    float*    pos   = (float*)(ws + off);    off += BDIM * sizeof(float);
    float*    rsum  = (float*)(ws + off);    off += BDIM * sizeof(float);
    float*    csum  = (float*)(ws + off);    off += BDIM * sizeof(float);

    prep_kernel<<<BDIM, 256, 0, stream>>>(x, y, xh, xl, yh, yl, xnorm, ynorm, pos);

    dim3 grid(BDIM / 256, BDIM / 128);
    // Row direction: dist[i][j] = ||xe_i - y_j||, diag = BIG
    dist_gemm<<<grid, 256, 0, stream>>>(xh, xl, yh, yl, xnorm, ynorm, dist);
    topk_hinge_row<<<BDIM, 256, 0, stream>>>(dist, pos, rsum);

    // Column direction: regenerate transposed (same values, roles swapped)
    dist_gemm<<<grid, 256, 0, stream>>>(yh, yl, xh, xl, ynorm, xnorm, dist);
    topk_hinge_row<<<BDIM, 256, 0, stream>>>(dist, pos, csum);

    finalize<<<1, 256, 0, stream>>>(rsum, csum, out);
}
